// ExtremeWeatherModel_60576218743379
// MI455X (gfx1250) — matrix operand order repaired and checked by reading, not compiled, not run on hardware
//
#include <hip/hip_runtime.h>
#include <hip/hip_bf16.h>

typedef __attribute__((ext_vector_type(16))) _Float16 v16h;
typedef __attribute__((ext_vector_type(8)))  _Float16 v8h;
typedef __attribute__((ext_vector_type(8)))  float    v8f;
typedef __attribute__((ext_vector_type(4)))  int      v4i;

#define HIDDEN 16
#define NB    8      // batch
#define NT    12     // time steps
#define CIN   5      // x channels
#define IMH   256
#define IMW   256
#define PSTR  24     // LDS halves per halo position per region (32B payload + 16B pad; 48B = 12-bank step, conflict-free)

#if __has_builtin(__builtin_amdgcn_global_load_async_to_lds_b128)
#define ASYNC_LDS 1
#else
#define ASYNC_LDS 0
#endif

#define GLOBAL_AS __attribute__((address_space(1)))
#define LDS_AS    __attribute__((address_space(3)))

__device__ __forceinline__ float sigm_f(float x)    { return 1.f / (1.f + __expf(-x)); }
__device__ __forceinline__ float tanh_fast(float x) { return 1.f - 2.f / (__expf(2.f * x) + 1.f); }

// ---------------------------------------------------------------------------
// Pre-swizzle W_cell [64][21][3][3] fp32 into f16 A-fragments (this code's
// contiguous-K 16x32 A layout, not the ISA table order): lane L holds row m=L%16; half j ->
// K = j + (L>=16 ? 16 : 0). Fragment id = tap*4 + gate-block.
// K ordering (matches LDS B staging): K 0..15 = h channels (W ic 5..20),
// K 16..20 = x channels (W ic 0..4), K 21..31 = zero pad.
// ---------------------------------------------------------------------------
__global__ void prep_weights(const float* __restrict__ Wc, _Float16* __restrict__ Af)
{
    int u = blockIdx.x * blockDim.x + threadIdx.x;
    if (u >= 36 * 32 * 16) return;
    int frag = u >> 9;
    int rem  = u & 511;
    int L    = rem >> 4;
    int j    = rem & 15;
    int tap  = frag >> 2;
    int ob   = frag & 3;
    int ky   = tap / 3, kx = tap % 3;
    int m    = L & 15;
    int K    = j + ((L >= 16) ? 16 : 0);
    int ic   = (K < HIDDEN) ? (CIN + K) : (K - HIDDEN);   // h first, then x
    float v  = 0.f;
    if (K < CIN + HIDDEN)
        v = Wc[(((size_t)(ob * 16 + m) * (CIN + HIDDEN) + ic) * 3 + ky) * 3 + kx];
    Af[u] = (_Float16)v;
}

// ---------------------------------------------------------------------------
// ConvLSTM cell step. Block = 256 threads = 8 waves; tile = 8 rows x 16 cols.
// h/c state is channel-last [b][y][x][16]: h halo staged via async
// global->LDS b128 copies (ASYNCcnt), x halo converted fp32->f16 by VALU.
// 9 taps x 4 gate blocks of v_wmma_f32_16x16x32_f16 (K padded 21->32).
// ---------------------------------------------------------------------------
__global__ __launch_bounds__(256)
void cell_step(const float* __restrict__ x,
               const _Float16* __restrict__ Af,
               const float* __restrict__ bcell,
               const _Float16* __restrict__ hprev,
               _Float16* __restrict__ hnext,
               float* __restrict__ cstate,
               int t, int first)
{
    // region 0: h channels (K 0..15), region 1: x channels + pad (K 16..31)
    __shared__ __align__(16) _Float16 bpatch[2][180 * PSTR];   // 2 x 8640 B

    const int tid  = threadIdx.x;
    const int b    = blockIdx.z;
    const int row0 = blockIdx.y * 8;
    const int col0 = blockIdx.x * 16;

    // ---- stage (10 x 18) halo patch ----
    if (tid < 180) {
        const int r  = tid / 18;
        const int cc = tid % 18;
        const int gy = row0 - 1 + r;
        const int gx = col0 - 1 + cc;
        const bool inb = (gy >= 0) & (gy < IMH) & (gx >= 0) & (gx < IMW);

        // x region: halves 0..4 = x channels, 5..15 = 0 (K zero pad)
        union { v8h v; _Float16 e[8]; } xa; xa.v = v8h{};
        #pragma unroll
        for (int c = 0; c < CIN; ++c) {
            float v = 0.f;
            if (inb) v = x[((((size_t)b * NT + t) * CIN + c) * IMH + gy) * IMW + gx];
            xa.e[c] = (_Float16)v;
        }
        _Float16* xdst = &bpatch[1][tid * PSTR];
        *(v8h*)(xdst)     = xa.v;
        *(v8h*)(xdst + 8) = v8h{};

        // h region: 16 contiguous halves per pixel in channel-last hprev
        _Float16* hdst = &bpatch[0][tid * PSTR];
        if (inb && !first) {
            const _Float16* hsrc = hprev + ((size_t)(b * IMH + gy) * IMW + gx) * HIDDEN;
#if ASYNC_LDS
            __builtin_amdgcn_global_load_async_to_lds_b128(
                (GLOBAL_AS v4i*)hsrc,
                (LDS_AS v4i*)hdst, 0, 0);
            __builtin_amdgcn_global_load_async_to_lds_b128(
                (GLOBAL_AS v4i*)(hsrc + 8),
                (LDS_AS v4i*)(hdst + 8), 0, 0);
#else
            *(v8h*)(hdst)     = *(const v8h*)(hsrc);
            *(v8h*)(hdst + 8) = *(const v8h*)(hsrc + 8);
#endif
        } else {
            *(v8h*)(hdst)     = v8h{};
            *(v8h*)(hdst + 8) = v8h{};
        }
    }
#if ASYNC_LDS
#if __has_builtin(__builtin_amdgcn_s_wait_asynccnt)
    __builtin_amdgcn_s_wait_asynccnt(0);
#else
    asm volatile("s_wait_asynccnt 0" ::: "memory");
#endif
#endif
    __syncthreads();

    const int wid  = tid >> 5;    // wave id = tile row 0..7
    const int lane = tid & 31;
    const int n    = lane & 15;   // pixel column / B column
    const int hi   = lane >> 4;   // K/M half selector (also picks LDS region)

    v8f acc[4] = {v8f{}, v8f{}, v8f{}, v8f{}};

    #pragma unroll
    for (int tap = 0; tap < 9; ++tap) {
        const int dy = tap / 3, dx = tap % 3;
        // B 32x16: lane = column n, halves j -> K = j + 16*hi.
        const _Float16* bp = &bpatch[hi][((wid + dy) * 18 + (n + dx)) * PSTR];
        union { v16h v; v8h h[2]; } bu;
        bu.h[0] = *(const v8h*)(bp);
        bu.h[1] = *(const v8h*)(bp + 8);
        #pragma unroll
        for (int ob = 0; ob < 4; ++ob) {
            const _Float16* ap = Af + (((size_t)(tap * 4 + ob) * 32 + lane) << 4);
            union { v16h v; v8h h[2]; } au;
            au.h[0] = *(const v8h*)(ap);
            au.h[1] = *(const v8h*)(ap + 8);
            acc[ob] = __builtin_amdgcn_wmma_f32_16x16x32_f16(
                false, au.v, false, bu.v, (short)0, acc[ob], false, false);
        }
    }

    // ---- LSTM gate epilogue (C/D layout: lane -> col n + M-half hi; vgpr v -> row)
    const int gy = row0 + wid;
    const int gx = col0 + n;
    const size_t chbase = ((size_t)(b * IMH + gy) * IMW + gx) * HIDDEN + hi * 8;

    v8f bi = *(const v8f*)(bcell +  0 + hi * 8);
    v8f bf = *(const v8f*)(bcell + 16 + hi * 8);
    v8f bo = *(const v8f*)(bcell + 32 + hi * 8);
    v8f bg = *(const v8f*)(bcell + 48 + hi * 8);
    v8f cold = v8f{};
    if (!first) cold = *(const v8f*)(cstate + chbase);

    v8f cnew;
    union { v8h v; _Float16 e[8]; } hout;
    #pragma unroll
    for (int v = 0; v < 8; ++v) {
        float ig = sigm_f(acc[0][v] + bi[v]);
        float fg = sigm_f(acc[1][v] + bf[v]);
        float og = sigm_f(acc[2][v] + bo[v]);
        float gg = tanh_fast(acc[3][v] + bg[v]);
        float cn = fg * cold[v] + ig * gg;
        cnew[v]  = cn;
        hout.e[v] = (_Float16)(og * tanh_fast(cn));
    }
    *(v8f*)(cstate + chbase) = cnew;
    *(v8h*)(hnext + chbase)  = hout.v;
}

// ---------------------------------------------------------------------------
// Final 1x1 conv over channel-last h: out[b,0,y,x] = b_final + sum_ch W*h
// ---------------------------------------------------------------------------
__global__ void final_conv(const _Float16* __restrict__ h,
                           const float* __restrict__ wf,
                           const float* __restrict__ bf,
                           float* __restrict__ out)
{
    int p = blockIdx.x * blockDim.x + threadIdx.x;
    if (p >= NB * IMH * IMW) return;
    const _Float16* hp = h + (size_t)p * HIDDEN;
    union { v8h v; _Float16 e[8]; } u0, u1;
    u0.v = *(const v8h*)(hp);
    u1.v = *(const v8h*)(hp + 8);
    float s = bf[0];
    #pragma unroll
    for (int c = 0; c < 8; ++c) s += wf[c] * (float)u0.e[c];
    #pragma unroll
    for (int c = 0; c < 8; ++c) s += wf[8 + c] * (float)u1.e[c];
    out[p] = s;
}

extern "C" void kernel_launch(void* const* d_in, const int* in_sizes, int n_in,
                              void* d_out, int out_size, void* d_ws, size_t ws_size,
                              hipStream_t stream)
{
    const float* x  = (const float*)d_in[0];   // [8,12,5,256,256]
    const float* Wc = (const float*)d_in[1];   // [64,21,3,3]
    const float* bc = (const float*)d_in[2];   // [64]
    const float* Wf = (const float*)d_in[3];   // [1,16,1,1]
    const float* bf = (const float*)d_in[4];   // [1]
    float* out = (float*)d_out;

    char* ws = (char*)d_ws;
    _Float16* Af = (_Float16*)ws;                                            // 36,864 B
    const size_t hbytes = (size_t)NB * IMH * IMW * HIDDEN * sizeof(_Float16); // 16 MB
    _Float16* h0 = (_Float16*)(ws + 65536);
    _Float16* h1 = (_Float16*)(ws + 65536 + hbytes);
    float*    cs = (float*)   (ws + 65536 + 2 * hbytes);                     // 32 MB

    prep_weights<<<dim3((36 * 32 * 16 + 255) / 256), 256, 0, stream>>>(Wc, Af);

    _Float16* hb[2] = {h0, h1};
    for (int t = 0; t < NT; ++t) {
        cell_step<<<dim3(IMW / 16, IMH / 8, NB), 256, 0, stream>>>(
            x, Af, bc, hb[t & 1], hb[(t + 1) & 1], cs, t, (t == 0) ? 1 : 0);
    }
    final_conv<<<dim3((NB * IMH * IMW + 255) / 256), 256, 0, stream>>>(hb[NT & 1], Wf, bf, out);
}
